// EGCL_22703197127077
// MI455X (gfx1250) — matrix operand order repaired and checked by reading, not compiled, not run on hardware
//
#include <hip/hip_runtime.h>
#include <hip/hip_bf16.h>
#include <math.h>

// ---------------- CDNA5 types ----------------
typedef _Float16 half_t;
typedef __attribute__((ext_vector_type(16))) _Float16 v16h;
typedef __attribute__((ext_vector_type(8)))  _Float16 v8h;
typedef __attribute__((ext_vector_type(8)))  float    v8f;
typedef __attribute__((ext_vector_type(4)))  float    v4f;
typedef __attribute__((ext_vector_type(4)))  unsigned v4u;
typedef __attribute__((ext_vector_type(8)))  int      v8i;
typedef __attribute__((ext_vector_type(4)))  int      v4i;

#define N_NODES 20000
#define N_EDGES 640000
#define HDIM    128
#define VDIM    3
#define DDIM    3

// LDS strides (halves); multiples of 8 so 16B vector ops stay aligned
#define FS 296   // edge feature rows: 288 used
#define MS 136   // 128-wide activation rows
#define IS 264   // node input rows: 256 used

// workspace layout (halves for packed f16 weights)
#define OFF_PEW0 0          // 9*8*512  = 36864
#define OFF_PEW1 36864      // 4*8*512  = 16384
#define OFF_PXW0 53248
#define OFF_PXW1 69632
#define OFF_XOI  86016      // 4*1*512  = 2048
#define OFF_PHW0 88064      // 8*8*512  = 32768
#define OFF_PHW1 120832
#define OFF_PHW2 137216
#define WS_W_BYTES 307200   // halves end 153600 -> bytes

// Tensor Data Mover availability (per-pass: false on host pass, probed on device)
#if defined(__has_builtin)
#  if __has_builtin(__builtin_amdgcn_tensor_load_to_lds)
#    define HAS_TDM 1
#  endif
#endif
#ifndef HAS_TDM
#  define HAS_TDM 0
#endif

__device__ inline void wave_lds_fence() {
  __builtin_amdgcn_wave_barrier();
  asm volatile("s_wait_dscnt 0" ::: "memory");
  __builtin_amdgcn_wave_barrier();
}

__device__ inline void atomF(float* p, float v) {
  __hip_atomic_fetch_add(p, v, __ATOMIC_RELAXED, __HIP_MEMORY_SCOPE_AGENT);
}

__device__ inline float silu(float x) { return x / (1.0f + __expf(-x)); }

// TDM: async copy of a contiguous 16x128 f32 tile (global -> LDS).
// D# per CDNA5 ISA: group0 = {count|flags, lds_addr,
// global_addr[56:0], type=2}, group1 packs data_size/dims/strides.
__device__ inline void tdm_load_tile16x128(const float* gsrc, unsigned ldsOff) {
#if HAS_TDM
  unsigned long long ga = (unsigned long long)(const void*)gsrc;
  v4u g0;
  g0[0] = 1u;                                        // count=1, user mode
  g0[1] = ldsOff;                                    // lds_addr (bytes)
  g0[2] = (unsigned)(ga & 0xFFFFFFFFu);              // global_addr[31:0]
  g0[3] = (unsigned)((ga >> 32) & 0x01FFFFFFu) | 0x80000000u;  // addr[56:32] | type=2
  v8i g1;
  g1[0] = 0x00020000;          // workgroup_mask=0, data_size=2 (4 bytes)
  g1[1] = (int)(128u << 16);   // tensor_dim0 = 128 (bits [79:48], low half)
  g1[2] = (int)(16u << 16);    // tensor_dim1 = 16  (bits [111:80], low half)
  g1[3] = (int)(128u << 16);   // tile_dim0 = 128   (bits [127:112])
  g1[4] = 16;                  // tile_dim1 = 16    (bits [143:128])
  g1[5] = 128;                 // tensor_dim0_stride = 128 (bits [207:160], low 32)
  g1[6] = 0;
  g1[7] = 0;
  v4i gz = {};
#if __clang_major__ >= 23
  v8i gz8 = {};
  __builtin_amdgcn_tensor_load_to_lds(g0, g1, gz, gz, gz8, 0);
#else
  __builtin_amdgcn_tensor_load_to_lds(g0, g1, gz, gz, 0);
#endif
#else
  (void)gsrc; (void)ldsOff;
#endif
}

// Build A fragment (16x32 f16) for WMMA from an LDS row.
// lane l: M = l&15 (row), hi = l>>4 selects K halves per ISA layout:
//   halves 0..7  -> K = 32c + 8*hi + (0..7)
//   halves 8..15 -> K = 32c + 16 + 8*hi + (0..7)
__device__ inline v16h mk_a(const half_t* row, int c, int hi) {
  const v8h lo = *(const v8h*)(row + 32 * c + 8 * hi);
  const v8h hh = *(const v8h*)(row + 32 * c + 16 + 8 * hi);
  v16h a;
#pragma unroll
  for (int i = 0; i < 8; ++i) { a[i] = lo[i]; a[8 + i] = hh[i]; }
  return a;
}

// Generic WMMA layer: LDS(f16, A layout rows) x packed-W -> LDS(f16)
// C/D layout: lane l, vgpr r holds (M = r + 8*(l>>4), N = (l&15) + 16*nt)
template <int KC, int NT, bool ACT>
__device__ inline void gemm_lds(const half_t* __restrict__ inBase, int inStride,
                                const half_t* __restrict__ W,
                                const float* __restrict__ bias,
                                half_t* __restrict__ outBase, int outStride,
                                int lane) {
  const int m  = lane & 15;
  const int hi = lane >> 4;
  const half_t* row = inBase + m * inStride;
  v8f acc[NT] = {};
#pragma unroll
  for (int c = 0; c < KC; ++c) {
    v16h a = mk_a(row, c, hi);
#pragma unroll
    for (int nt = 0; nt < NT; ++nt) {
      v16h b = *(const v16h*)(W + ((size_t)(c * NT + nt) * 32 + lane) * 16);
      acc[nt] = __builtin_amdgcn_wmma_f32_16x16x32_f16(
          false, a, false, b, (short)0, acc[nt], false, false);
    }
  }
#pragma unroll
  for (int nt = 0; nt < NT; ++nt) {
    float bs = bias[nt * 16 + m];
#pragma unroll
    for (int r = 0; r < 8; ++r) {
      float v = acc[nt][r] + bs;
      if (ACT) v = silu(v);
      outBase[(r + 8 * hi) * outStride + m + 16 * nt] = (half_t)v;
    }
  }
  wave_lds_fence();
}

// ---------------- weight packing ----------------
// Fragment layout: dst[frag*512 + lane*16 + j]
// B element (k,n): lane = (n&15) + 16*((k>>3)&1) within chunk, j = (k&7) + 8*((k>>4)&1)
__global__ __launch_bounds__(256) void pack_w(const float* __restrict__ src,
                                              int Kreal, int Nreal, int KC, int NT,
                                              half_t* __restrict__ dst) {
  int idx = blockIdx.x * 256 + threadIdx.x;
  int total = KC * NT * 512;
  if (idx >= total) return;
  int frag = idx >> 9, within = idx & 511;
  int lane = within >> 4, j = within & 15;
  int kc = frag / NT, nt = frag % NT;
  int n = nt * 16 + (lane & 15);
  int k = kc * 32 + (j & 7) + 8 * (lane >> 4) + 16 * (j >> 3);
  float v = (k < Kreal && n < Nreal) ? src[(size_t)k * Nreal + n] : 0.0f;
  dst[idx] = (half_t)v;
}

// fused phi_x_out (128x3) + phi_inf (128x1) -> 128x16 tile (cols 0..2 | 3)
__global__ __launch_bounds__(256) void pack_xoi(const float* __restrict__ xw,
                                                const float* __restrict__ iw,
                                                half_t* __restrict__ dst) {
  int idx = blockIdx.x * 256 + threadIdx.x;
  if (idx >= 4 * 512) return;
  int frag = idx >> 9, within = idx & 511;
  int lane = within >> 4, j = within & 15;
  int n = lane & 15;
  int k = frag * 32 + (j & 7) + 8 * (lane >> 4) + 16 * (j >> 3);
  float v = 0.0f;
  if (n < 3)       v = xw[(size_t)k * 3 + n];
  else if (n == 3) v = iw[k];
  dst[idx] = (half_t)v;
}

__global__ __launch_bounds__(256) void init_kernel(const float* __restrict__ pos,
                                                   float* __restrict__ outv,
                                                   float* __restrict__ mi) {
  size_t idx = (size_t)blockIdx.x * 256 + threadIdx.x;
  if (idx < (size_t)N_NODES * HDIM) mi[idx] = 0.0f;
  if (idx < (size_t)N_NODES * VDIM * DDIM) outv[idx] = pos[idx];
}

// ---------------- edge kernel: 4 waves/block, 16 edges/wave ----------------
__global__ __launch_bounds__(128) void edge_kernel(
    const float* __restrict__ pos, const float* __restrict__ feats,
    const int* __restrict__ snd, const int* __restrict__ rcv,
    const half_t* __restrict__ w0, const float* __restrict__ b0,
    const half_t* __restrict__ w1, const float* __restrict__ b1,
    const half_t* __restrict__ xw0, const float* __restrict__ xb0,
    const half_t* __restrict__ xw1, const float* __restrict__ xb1,
    const half_t* __restrict__ xoi, const float* __restrict__ xob,
    const float* __restrict__ ifb,
    float* __restrict__ outv, float* __restrict__ mi) {
  __shared__ __align__(16) half_t s_feat[4][16 * FS];
  __shared__ __align__(16) half_t s_mij[4][16 * MS];
  __shared__ float s_small[4][64];
  __shared__ int   s_recv[4][16];

  const int wave = threadIdx.x >> 5;
  const int lane = threadIdx.x & 31;
  const int m = lane & 15, hi = lane >> 4;
  const long tile = (long)blockIdx.x * 4 + wave;
  const long e0 = tile * 16;
  if (e0 >= N_EDGES) return;

  half_t* feat = s_feat[wave];
  half_t* mij  = s_mij[wave];
  half_t* ping = s_feat[wave];          // reuse after L0 consumed
  half_t* pong = s_feat[wave] + 2304;
  float*  sml  = s_small[wave];
  int*    srcv = s_recv[wave];

  __builtin_prefetch(w0, 0, 3);

  // ---- stage edge features: lane hi=0 -> sender row (cols 0..127),
  //      lane hi=1 -> receiver row (cols 128..255)
  const long eIdx = e0 + m;
  const int node = hi ? rcv[eIdx] : snd[eIdx];
  if (hi) srcv[m] = node;
  {
    const float* src = feats + (size_t)node * HDIM;
    half_t* dst = feat + m * FS + hi * 128;
#pragma unroll
    for (int j = 0; j < 16; ++j) {
      v4f p0 = *(const v4f*)(src + j * 8);
      v4f p1 = *(const v4f*)(src + j * 8 + 4);
      v8h h;
#pragma unroll
      for (int i = 0; i < 4; ++i) { h[i] = (half_t)p0[i]; h[4 + i] = (half_t)p1[i]; }
      *(v8h*)(dst + j * 8) = h;
    }
  }

  // ---- geometry (lanes hi==0 own edge m): vectors, lengths, sq_lengths
  float vec[9];
  float len3[3];
  if (hi == 0) {
    const int rn = rcv[eIdx];
    const float* ps = pos + (size_t)node * 9;
    const float* pr = pos + (size_t)rn * 9;
    v8h s0 = {};
    v8h z8 = {};
#pragma unroll
    for (int v = 0; v < 3; ++v) {
      float n2 = 0.0f;
#pragma unroll
      for (int d = 0; d < 3; ++d) {
        float dv = pr[v * 3 + d] - ps[v * 3 + d];
        vec[v * 3 + d] = dv;
        n2 += dv * dv;
      }
      len3[v] = sqrtf(n2);
      s0[v] = (half_t)n2;
    }
    half_t* rowp = feat + m * FS;
    *(v8h*)(rowp + 256) = s0;
    *(v8h*)(rowp + 264) = z8;
    *(v8h*)(rowp + 272) = z8;
    *(v8h*)(rowp + 280) = z8;
  }
  wave_lds_fence();

  // ---- phi_e: 259(->288) -> 128 -> 128  (m_ij in s_mij)
  gemm_lds<9, 8, true>(feat, FS, w0, b0, mij, MS, lane);
  gemm_lds<4, 8, true>(mij, MS, w1, b1, mij, MS, lane);

  // ---- phi_x torso: 128 -> 128 -> 128
  gemm_lds<4, 8, true>(mij, MS, xw0, xb0, ping, MS, lane);
  gemm_lds<4, 8, true>(ping, MS, xw1, xb1, pong, MS, lane);

  // ---- fused phi_x_out (cols 0..2) + phi_inf sigmoid gate (col 3)
  {
    const half_t* row = pong + m * MS;
    v8f acc = {};
#pragma unroll
    for (int c = 0; c < 4; ++c) {
      v16h a = mk_a(row, c, hi);
      v16h b = *(const v16h*)(xoi + ((size_t)c * 32 + lane) * 16);
      acc = __builtin_amdgcn_wmma_f32_16x16x32_f16(
          false, a, false, b, (short)0, acc, false, false);
    }
    const int col = lane & 15;
    if (col < 4) {
      float bs = (col < 3) ? xob[col] : ifb[0];
#pragma unroll
      for (int r = 0; r < 8; ++r) {
        float v = acc[r] + bs;
        if (col == 3) v = 1.0f / (1.0f + __expf(-v));
        sml[(r + 8 * hi) * 4 + col] = v;
      }
    }
  }
  wave_lds_fence();

  // ---- equivariant shifts: atomics into vectors_out (seeded with positions)
  if (hi == 0) {
    const int rn = rcv[eIdx];
    const float inv = 1.0f / (float)(N_NODES - 1);
#pragma unroll
    for (int v = 0; v < 3; ++v) {
      float sc = sml[m * 4 + v] / (1.0f + len3[v]) * inv;
#pragma unroll
      for (int d = 0; d < 3; ++d)
        atomF(&outv[(size_t)rn * 9 + v * 3 + d], sc * vec[v * 3 + d]);
    }
  }

  // ---- gated scalar aggregation: m_i += m_ij * e  (4 channels per lane)
  {
    const int ch = lane * 4;
#pragma unroll 4
    for (int mm = 0; mm < 16; ++mm) {
      float eg = sml[mm * 4 + 3];
      float* dstp = mi + (size_t)srcv[mm] * HDIM + ch;
#pragma unroll
      for (int q = 0; q < 4; ++q)
        atomF(&dstp[q], (float)mij[mm * MS + ch + q] * eg);
    }
  }
}

// ---------------- node kernel: phi_h, 2 waves/block, 16 nodes/wave ----------
// Inputs are contiguous 16x128 f32 tiles -> fetched with the Tensor Data Mover
// (async, TENSORcnt) into LDS scratch, then converted f32->f16 into A-rows.
__global__ __launch_bounds__(64) void node_kernel(
    const float* __restrict__ feats, const float* __restrict__ mi,
    const half_t* __restrict__ hw0, const float* __restrict__ hb0,
    const half_t* __restrict__ hw1, const float* __restrict__ hb1,
    const half_t* __restrict__ hw2, const float* __restrict__ hb2,
    float* __restrict__ outf) {
  __shared__ __align__(16) float  n_scr[2][2][16 * 128];  // [wave][mi|feats] f32 tiles
  __shared__ __align__(16) half_t n_in[2][16 * IS];
  __shared__ __align__(16) half_t n_b[2][16 * MS];

  const int wave = threadIdx.x >> 5;
  const int lane = threadIdx.x & 31;
  const int m = lane & 15, hi = lane >> 4;
  const long tile = (long)blockIdx.x * 2 + wave;
  const long n0 = tile * 16;
  if (n0 >= N_NODES) return;

  half_t* inB = n_in[wave];
  half_t* bB  = n_b[wave];
  half_t* cB  = n_in[wave];  // input region dead after L0

  // stage concat(m_i/sqrt(avg), h): hi=0 -> m_i cols 0..127, hi=1 -> h cols 128..255
  {
    const float scl = hi ? 1.0f : rsqrtf((float)(N_NODES - 1));
#if HAS_TDM
    tdm_load_tile16x128(mi + (size_t)n0 * HDIM,
                        (unsigned)(size_t)(void*)&n_scr[wave][0][0]);
    tdm_load_tile16x128(feats + (size_t)n0 * HDIM,
                        (unsigned)(size_t)(void*)&n_scr[wave][1][0]);
    __builtin_amdgcn_s_wait_tensorcnt(0);
    __builtin_amdgcn_wave_barrier();
    const float* src = &n_scr[wave][hi][m * 128];
#else
    const long node = n0 + m;
    const float* src = hi ? (feats + (size_t)node * HDIM) : (mi + (size_t)node * HDIM);
#endif
    half_t* dst = inB + m * IS + hi * 128;
#pragma unroll
    for (int j = 0; j < 16; ++j) {
      v4f p0 = *(const v4f*)(src + j * 8);
      v4f p1 = *(const v4f*)(src + j * 8 + 4);
      v8h h;
#pragma unroll
      for (int i = 0; i < 4; ++i) { h[i] = (half_t)(p0[i] * scl); h[4 + i] = (half_t)(p1[i] * scl); }
      *(v8h*)(dst + j * 8) = h;
    }
  }
  wave_lds_fence();

  gemm_lds<8, 8, true>(inB, IS, hw0, hb0, bB, MS, lane);
  gemm_lds<4, 8, true>(bB, MS, hw1, hb1, cB, MS, lane);

  // final layer: no activation, + residual h, store f32
  {
    const half_t* row = cB + m * MS;
    v8f acc[8] = {};
#pragma unroll
    for (int c = 0; c < 4; ++c) {
      v16h a = mk_a(row, c, hi);
#pragma unroll
      for (int nt = 0; nt < 8; ++nt) {
        v16h b = *(const v16h*)(hw2 + ((size_t)(c * 8 + nt) * 32 + lane) * 16);
        acc[nt] = __builtin_amdgcn_wmma_f32_16x16x32_f16(
            false, a, false, b, (short)0, acc[nt], false, false);
      }
    }
    const int col = lane & 15;
#pragma unroll
    for (int nt = 0; nt < 8; ++nt) {
      const int n = nt * 16 + col;
      float bs = hb2[n];
#pragma unroll
      for (int r = 0; r < 8; ++r) {
        const long node = n0 + r + 8 * hi;
        float v = acc[nt][r] + bs + feats[(size_t)node * HDIM + n];
        outf[(size_t)node * HDIM + n] = v;
      }
    }
  }
}

// ---------------- launcher ----------------
extern "C" void kernel_launch(void* const* d_in, const int* in_sizes, int n_in,
                              void* d_out, int out_size, void* d_ws, size_t ws_size,
                              hipStream_t stream) {
  const float* pos   = (const float*)d_in[0];
  const float* feats = (const float*)d_in[1];
  const int*   snd   = (const int*)d_in[2];
  const int*   rcv   = (const int*)d_in[3];

  half_t* wsh = (half_t*)d_ws;
  half_t* pe_w0 = wsh + OFF_PEW0;
  half_t* pe_w1 = wsh + OFF_PEW1;
  half_t* px_w0 = wsh + OFF_PXW0;
  half_t* px_w1 = wsh + OFF_PXW1;
  half_t* xoi   = wsh + OFF_XOI;
  half_t* ph_w0 = wsh + OFF_PHW0;
  half_t* ph_w1 = wsh + OFF_PHW1;
  half_t* ph_w2 = wsh + OFF_PHW2;
  float*  mi    = (float*)((char*)d_ws + WS_W_BYTES);

  float* outv = (float*)d_out;
  float* outf = outv + (size_t)N_NODES * VDIM * DDIM;

  // pack weights -> f16 WMMA fragment layout
  pack_w<<<(9 * 8 * 512 + 255) / 256, 256, 0, stream>>>((const float*)d_in[4], 259, 128, 9, 8, pe_w0);
  pack_w<<<(4 * 8 * 512 + 255) / 256, 256, 0, stream>>>((const float*)d_in[6], 128, 128, 4, 8, pe_w1);
  pack_w<<<(4 * 8 * 512 + 255) / 256, 256, 0, stream>>>((const float*)d_in[8], 128, 128, 4, 8, px_w0);
  pack_w<<<(4 * 8 * 512 + 255) / 256, 256, 0, stream>>>((const float*)d_in[10], 128, 128, 4, 8, px_w1);
  pack_xoi<<<(4 * 512 + 255) / 256, 256, 0, stream>>>((const float*)d_in[12], (const float*)d_in[14], xoi);
  pack_w<<<(8 * 8 * 512 + 255) / 256, 256, 0, stream>>>((const float*)d_in[16], 256, 128, 8, 8, ph_w0);
  pack_w<<<(4 * 8 * 512 + 255) / 256, 256, 0, stream>>>((const float*)d_in[18], 128, 128, 4, 8, ph_w1);
  pack_w<<<(4 * 8 * 512 + 255) / 256, 256, 0, stream>>>((const float*)d_in[20], 128, 128, 4, 8, ph_w2);

  // init m_i accumulator + seed vectors_out with positions (residual)
  init_kernel<<<((size_t)N_NODES * HDIM + 255) / 256, 256, 0, stream>>>(pos, outv, mi);

  // per-edge MLPs + scatter
  edge_kernel<<<N_EDGES / 64, 128, 0, stream>>>(
      pos, feats, snd, rcv,
      pe_w0, (const float*)d_in[5], pe_w1, (const float*)d_in[7],
      px_w0, (const float*)d_in[9], px_w1, (const float*)d_in[11],
      xoi, (const float*)d_in[13], (const float*)d_in[15],
      outv, mi);

  // per-node phi_h (TDM-staged inputs)
  node_kernel<<<(N_NODES / 16) / 2, 64, 0, stream>>>(
      feats, mi,
      ph_w0, (const float*)d_in[17], ph_w1, (const float*)d_in[19],
      ph_w2, (const float*)d_in[21], outf);
}
